// SigGraInferNet_GAT_75849122448215
// MI455X (gfx1250) — matrix operand order repaired and checked by reading, not compiled, not run on hardware
//
#include <hip/hip_runtime.h>

// ---------------- problem constants (from reference) ----------------
#define BT 4
#define NN 2048
#define FIN 512
#define FOUT 128
#define PIN 256
#define POUT 128
#define HEAD 4
#define KD 32            // per-head dim
#define NEG_SLOPE 0.2f
#define MASK_VAL (-1e30f)

typedef __attribute__((ext_vector_type(16))) __bf16 v16bf;
typedef __attribute__((ext_vector_type(8)))  float  v8f;
typedef __attribute__((ext_vector_type(4)))  unsigned int v4u;
typedef __attribute__((ext_vector_type(8)))  int    v8i;
typedef __attribute__((ext_vector_type(4)))  int    v4i;

// ---- Tensor Data Mover availability (ROCm7.2 clang-22: 5 args; clang-23: 6) ----
#if defined(__has_builtin)
#if __has_builtin(__builtin_amdgcn_tensor_load_to_lds) && __has_builtin(__builtin_amdgcn_s_wait_tensorcnt)
#define USE_TDM 1
#endif
#endif
#ifndef USE_TDM
#define USE_TDM 0
#endif

#if USE_TDM
#if __clang_major__ >= 23
#define TDM_LOAD(g0, g1, g2, g3) \
    __builtin_amdgcn_tensor_load_to_lds(g0, g1, g2, g3, (v8i){0,0,0,0,0,0,0,0}, 0)
#else
#define TDM_LOAD(g0, g1, g2, g3) \
    __builtin_amdgcn_tensor_load_to_lds(g0, g1, g2, g3, 0)
#endif

// Issue a TDM 2D tile load: 16 rows x 32 cols of int32 from a row-major
// [*,2048] tensor into LDS (contiguous row-major 16x32 tile).
static __device__ __forceinline__ void tdm_load_adj_tile(const int* gptr, unsigned lds_off) {
    const unsigned long long ga = (unsigned long long)(uintptr_t)gptr;
    // D# group0: count=1 | lds_addr | global_addr[56:0] | type=2
    v4u g0 = { 1u,
               lds_off,
               (unsigned)ga,
               (unsigned)((ga >> 32) & 0x01FFFFFFu) | (2u << 30) };
    // D# group1: data_size=4B (code2); tensor_dim0=2048; tensor_dim1=2048;
    // tile_dim0=32; tile_dim1=16; tensor_dim0_stride=2048 elements.
    v8i g1 = { (int)(2u << 16),
               (int)(2048u << 16),          // tensor_dim0[15:0] @ bits 63:48
               (int)(2048u << 16),          // tensor_dim0[31:16]=0 | tensor_dim1[15:0] @ 95:80
               (int)(32u << 16),            // tensor_dim1[31:16]=0 | tile_dim0 @ 127:112
               16,                          // tile_dim1
               2048,                        // tensor_dim0_stride[31:0]
               0, 0 };
    v4i gz = { 0, 0, 0, 0 };
    TDM_LOAD(g0, g1, gz, gz);
}
#endif

static __device__ __forceinline__ float lrelu(float x) {
    return x >= 0.f ? x : NEG_SLOPE * x;
}

// =====================================================================
// Generic Y[M,Nout] = (X[M,Kd] @ W[Nout,Kd]^T) + bias, bf16 WMMA, f32 acc.
// Block = 128 threads (4 waves); each wave produces one 16x16 tile.
// Grid: (M/16, Nout/64).
// =====================================================================
__global__ void gemm_bf16_wmma(const float* __restrict__ X,
                               const float* __restrict__ W,
                               const float* __restrict__ bias,
                               float* __restrict__ Y,
                               int M, int Kd, int Nout, int relu) {
    const int wave = threadIdx.x >> 5;
    const int lane = threadIdx.x & 31;
    const int i0 = blockIdx.x * 16;
    const int n0 = blockIdx.y * 64 + wave * 16;
    const int m  = lane & 15;
    const int hh = lane >> 4;
    const int nc = lane & 15;

    const float bv = bias ? bias[n0 + nc] : 0.f;    // hoisted (no per-element branch)

    v8f acc = {0.f,0.f,0.f,0.f,0.f,0.f,0.f,0.f};
    for (int k0 = 0; k0 < Kd; k0 += 32) {
        v16bf afrag, bfrag;
        // A 16x32 bf16: lane m, K(e) = e + 8*hh + 8*(e>>3) -> two 8-float runs
        {
            const float* xr = X + (size_t)(i0 + m) * Kd + k0 + 8 * hh;
            const float4 a0 = *(const float4*)(xr);
            const float4 a1 = *(const float4*)(xr + 4);
            const float4 a2 = *(const float4*)(xr + 16);
            const float4 a3 = *(const float4*)(xr + 20);
            afrag[0]=(__bf16)a0.x; afrag[1]=(__bf16)a0.y; afrag[2]=(__bf16)a0.z; afrag[3]=(__bf16)a0.w;
            afrag[4]=(__bf16)a1.x; afrag[5]=(__bf16)a1.y; afrag[6]=(__bf16)a1.z; afrag[7]=(__bf16)a1.w;
            afrag[8]=(__bf16)a2.x; afrag[9]=(__bf16)a2.y; afrag[10]=(__bf16)a2.z; afrag[11]=(__bf16)a2.w;
            afrag[12]=(__bf16)a3.x; afrag[13]=(__bf16)a3.y; afrag[14]=(__bf16)a3.z; afrag[15]=(__bf16)a3.w;
        }
        // B 32x16 bf16: lane n, K(e) = e + 8*hh + 8*(e>>3) ; B[k][n] = W[n][k] -> two 8-float runs
        {
            const float* wr = W + (size_t)(n0 + nc) * Kd + k0 + 8 * hh;
            const float4 b0 = *(const float4*)(wr);
            const float4 b1 = *(const float4*)(wr + 4);
            const float4 b2 = *(const float4*)(wr + 16);
            const float4 b3 = *(const float4*)(wr + 20);
            bfrag[0]=(__bf16)b0.x; bfrag[1]=(__bf16)b0.y; bfrag[2]=(__bf16)b0.z; bfrag[3]=(__bf16)b0.w;
            bfrag[4]=(__bf16)b1.x; bfrag[5]=(__bf16)b1.y; bfrag[6]=(__bf16)b1.z; bfrag[7]=(__bf16)b1.w;
            bfrag[8]=(__bf16)b2.x; bfrag[9]=(__bf16)b2.y; bfrag[10]=(__bf16)b2.z; bfrag[11]=(__bf16)b2.w;
            bfrag[12]=(__bf16)b3.x; bfrag[13]=(__bf16)b3.y; bfrag[14]=(__bf16)b3.z; bfrag[15]=(__bf16)b3.w;
        }
        acc = __builtin_amdgcn_wmma_f32_16x16x32_bf16(
            false, afrag, false, bfrag, (short)0, acc, false, false);
    }
#pragma unroll
    for (int e = 0; e < 8; ++e) {
        const int mm = e + 8 * hh;
        float v = acc[e] + bv;
        if (relu) v = fmaxf(v, 0.f);
        Y[(size_t)(i0 + mm) * Nout + n0 + nc] = v;
    }
}

// =====================================================================
// s1[b,h,n] = xh[b,n,h,:].attw[h,:K] ; s2 = .attw[h,K:]
// =====================================================================
__global__ void s12_kernel(const float* __restrict__ xh,
                           const float* __restrict__ attw,
                           float* __restrict__ s1,
                           float* __restrict__ s2) {
    const int tid = blockIdx.x * blockDim.x + threadIdx.x;  // b*H*N + h*N + n
    if (tid >= BT * HEAD * NN) return;
    const int n = tid % NN;
    const int h = (tid / NN) % HEAD;
    const int b = tid / (HEAD * NN);
    const float* xp = xh + ((size_t)(b * NN + n)) * POUT + h * KD;
    float a1 = 0.f, a2 = 0.f;
#pragma unroll
    for (int k = 0; k < KD; k += 4) {
        const float4 xv = *(const float4*)(xp + k);
        const float4 w1 = *(const float4*)(attw + h * 2 * KD + k);
        const float4 w2 = *(const float4*)(attw + h * 2 * KD + KD + k);
        a1 += xv.x * w1.x + xv.y * w1.y + xv.z * w1.z + xv.w * w1.w;
        a2 += xv.x * w2.x + xv.y * w2.y + xv.z * w2.z + xv.w * w2.w;
    }
    s1[tid] = a1;
    s2[tid] = a2;
}

// =====================================================================
// Flash pass 1: per row (b,h,i): mx = max_j score, il = 1/sum exp.
// One wave per row; int4/float4 streaming; wave32 shuffle reductions.
// =====================================================================
__global__ void rowstats_kernel(const int* __restrict__ Aadj,
                                const float* __restrict__ s1,
                                const float* __restrict__ s2,
                                const float* __restrict__ attb,
                                float* __restrict__ mx,
                                float* __restrict__ il) {
    const int lane = threadIdx.x & 31;
    const int wave = threadIdx.x >> 5;
    const int row  = blockIdx.x * (blockDim.x >> 5) + wave;  // b*H*N + h*N + i
    if (row >= BT * HEAD * NN) return;
    const int i = row % NN;
    const int h = (row / NN) % HEAD;
    const int b = row / (HEAD * NN);
    const float s1v = s1[row] + attb[h];
    const int*   arow = Aadj + ((size_t)(b * NN + i)) * NN;
    const float* s2r  = s2 + (size_t)(b * HEAD + h) * NN;

    float m = MASK_VAL;
    for (int j4 = lane * 4; j4 < NN; j4 += 128) {
        const int4   av = *(const int4*)(arow + j4);
        const float4 sv = *(const float4*)(s2r + j4);
        m = fmaxf(m, (av.x > 0) ? lrelu(s1v + sv.x) : MASK_VAL);
        m = fmaxf(m, (av.y > 0) ? lrelu(s1v + sv.y) : MASK_VAL);
        m = fmaxf(m, (av.z > 0) ? lrelu(s1v + sv.z) : MASK_VAL);
        m = fmaxf(m, (av.w > 0) ? lrelu(s1v + sv.w) : MASK_VAL);
    }
#pragma unroll
    for (int off = 16; off > 0; off >>= 1)
        m = fmaxf(m, __shfl_xor(m, off, 32));

    float s = 0.f;
    for (int j4 = lane * 4; j4 < NN; j4 += 128) {
        const int4   av = *(const int4*)(arow + j4);
        const float4 sv = *(const float4*)(s2r + j4);
        s += __expf(((av.x > 0) ? lrelu(s1v + sv.x) : MASK_VAL) - m);
        s += __expf(((av.y > 0) ? lrelu(s1v + sv.y) : MASK_VAL) - m);
        s += __expf(((av.z > 0) ? lrelu(s1v + sv.z) : MASK_VAL) - m);
        s += __expf(((av.w > 0) ? lrelu(s1v + sv.w) : MASK_VAL) - m);
    }
#pragma unroll
    for (int off = 16; off > 0; off >>= 1)
        s += __shfl_xor(s, off, 32);

    if (lane == 0) { mx[row] = m; il[row] = 1.f / s; }
}

// =====================================================================
// Flash pass 2 (WMMA core): G[b,i,h*32+d] = relu(sum_j p_ij xh[b,j,h*32+d]) + x
// Adjacency tiles double-buffered into LDS by the Tensor Data Mover;
// p regenerated on the fly into bf16 A-fragment layout.
// Grid: BT*HEAD*(NN/16) blocks of 128 (4 waves splitting j).
// =====================================================================
__global__ void attn_apply_kernel(const int* __restrict__ Aadj,
                                  const float* __restrict__ s1,
                                  const float* __restrict__ s2,
                                  const float* __restrict__ attb,
                                  const float* __restrict__ mx,
                                  const float* __restrict__ il,
                                  const float* __restrict__ xh,
                                  const float* __restrict__ xres,
                                  float* __restrict__ G) {
    __shared__ float s2s[NN];                       // staged s2 row (8KB)
    __shared__ float red[4][16][32];                // cross-wave partials (8KB)
    __shared__ float s1s[16], mxs[16], ils[16];
#if USE_TDM
    __shared__ __align__(16) int atile[4][2][16][32];  // per-wave double buffer (16KB)
#endif

    const int it = blockIdx.x & 127;
    const int h  = (blockIdx.x >> 7) & 3;
    const int b  = blockIdx.x >> 9;
    const int i0 = it * 16;
    const int lane = threadIdx.x & 31;
    const int wave = threadIdx.x >> 5;
    const int rowbase = (b * HEAD + h) * NN + i0;

    const float* s2r = s2 + (size_t)(b * HEAD + h) * NN;
    for (int j = threadIdx.x; j < NN; j += 128) s2s[j] = s2r[j];
    if (threadIdx.x < 16) {
        s1s[threadIdx.x] = s1[rowbase + threadIdx.x] + attb[h];
        mxs[threadIdx.x] = mx[rowbase + threadIdx.x];
        ils[threadIdx.x] = il[rowbase + threadIdx.x];
    }
    __syncthreads();

    const int m  = lane & 15;
    const int hh = lane >> 4;
    const int nc = lane & 15;
    const float s1m = s1s[m], mxm = mxs[m], ilm = ils[m];
    const int* abase = Aadj + ((size_t)(b * NN + i0)) * NN;   // 16-row block base

    v8f acc0 = {0.f,0.f,0.f,0.f,0.f,0.f,0.f,0.f};
    v8f acc1 = {0.f,0.f,0.f,0.f,0.f,0.f,0.f,0.f};

    const int jbeg = wave * 512;
#if USE_TDM
    const unsigned lds0 = (unsigned)(uintptr_t)&atile[wave][0][0][0];
    const unsigned lds1 = (unsigned)(uintptr_t)&atile[wave][1][0][0];
    tdm_load_adj_tile(abase + jbeg, lds0);          // prologue: tile 0 in flight
#else
    const int* arow = abase + (size_t)m * NN;
#endif

    for (int c = 0; c < 16; ++c) {
        const int j0 = jbeg + c * 32;
        int adjv[16];
#if USE_TDM
        if (c + 1 < 16) {
            // protect the buffer we are about to overwrite from in-flight LDS reads
            asm volatile("s_wait_dscnt 0x0" ::: "memory");
            tdm_load_adj_tile(abase + j0 + 32, (c & 1) ? lds0 : lds1);
            __builtin_amdgcn_s_wait_tensorcnt(1);   // tile c resident (in-order TDM)
        } else {
            __builtin_amdgcn_s_wait_tensorcnt(0);
        }
        asm volatile("" ::: "memory");
        {
            const int* arow_l = &atile[wave][c & 1][m][0];
            const int4 q0 = *(const int4*)(arow_l + 8 * hh);
            const int4 q1 = *(const int4*)(arow_l + 8 * hh + 4);
            const int4 q2 = *(const int4*)(arow_l + 16 + 8 * hh);
            const int4 q3 = *(const int4*)(arow_l + 16 + 8 * hh + 4);
            adjv[0]=q0.x; adjv[1]=q0.y; adjv[2]=q0.z; adjv[3]=q0.w;
            adjv[4]=q1.x; adjv[5]=q1.y; adjv[6]=q1.z; adjv[7]=q1.w;
            adjv[8]=q2.x; adjv[9]=q2.y; adjv[10]=q2.z; adjv[11]=q2.w;
            adjv[12]=q3.x; adjv[13]=q3.y; adjv[14]=q3.z; adjv[15]=q3.w;
        }
#else
        __builtin_prefetch(arow + j0 + 32, 0, 1);
        {
            const int* ap = arow + j0 + 8 * hh;
            const int4 q0 = *(const int4*)(ap);
            const int4 q1 = *(const int4*)(ap + 4);
            const int4 q2 = *(const int4*)(ap + 16);
            const int4 q3 = *(const int4*)(ap + 20);
            adjv[0]=q0.x; adjv[1]=q0.y; adjv[2]=q0.z; adjv[3]=q0.w;
            adjv[4]=q1.x; adjv[5]=q1.y; adjv[6]=q1.z; adjv[7]=q1.w;
            adjv[8]=q2.x; adjv[9]=q2.y; adjv[10]=q2.z; adjv[11]=q2.w;
            adjv[12]=q3.x; adjv[13]=q3.y; adjv[14]=q3.z; adjv[15]=q3.w;
        }
#endif
        v16bf afrag, bf0, bf1;
        // A-frag: p tile 16(i) x 32(j), K(e) = e + 8*hh + 8*(e>>3)
#pragma unroll
        for (int e = 0; e < 16; ++e) {
            const int j = j0 + e + 8 * hh + ((e >> 3) << 3);
            const float sc = (adjv[e] > 0) ? lrelu(s1m + s2s[j]) : MASK_VAL;
            afrag[e] = (__bf16)(__expf(sc - mxm) * ilm);
        }
        // B-frags: xh tile 32(j) x 16(d), two d-halves; B[k][n], k = j0 + e + 8*hh + 8*(e>>3)
#pragma unroll
        for (int e = 0; e < 16; ++e) {
            const float* xp = xh + ((size_t)(b * NN + j0 + e + 8 * hh + ((e >> 3) << 3))) * POUT + h * KD;
            bf0[e] = (__bf16)xp[nc];
            bf1[e] = (__bf16)xp[16 + nc];
        }
        acc0 = __builtin_amdgcn_wmma_f32_16x16x32_bf16(
            false, afrag, false, bf0, (short)0, acc0, false, false);
        acc1 = __builtin_amdgcn_wmma_f32_16x16x32_bf16(
            false, afrag, false, bf1, (short)0, acc1, false, false);
    }
    // D layout: lane n = l&15, M(e) = e + 8*hh
#pragma unroll
    for (int e = 0; e < 8; ++e) {
        red[wave][e + 8 * hh][nc]      = acc0[e];
        red[wave][e + 8 * hh][16 + nc] = acc1[e];
    }
    __syncthreads();
    for (int idx = threadIdx.x; idx < 512; idx += 128) {
        const int mm = idx >> 5;
        const int d  = idx & 31;
        const float sum = red[0][mm][d] + red[1][mm][d] + red[2][mm][d] + red[3][mm][d];
        const size_t o = ((size_t)(b * NN + i0 + mm)) * POUT + h * KD + d;
        G[o] = fmaxf(sum, 0.f) + xres[o];
    }
}

// =====================================================================
// Tiny fp32 GEMM (BT=4-row matmuls): one thread per output element.
// =====================================================================
__global__ void gemm_small(const float* __restrict__ X,
                           const float* __restrict__ W,
                           const float* __restrict__ bias,
                           float* __restrict__ Y,
                           int M, int Kd, int Nout, int relu) {
    const int tid = blockIdx.x * blockDim.x + threadIdx.x;
    if (tid >= M * Nout) return;
    const int mr = tid / Nout, n = tid % Nout;
    float acc = bias ? bias[n] : 0.f;
    for (int k = 0; k < Kd; ++k) acc += X[(size_t)mr * Kd + k] * W[(size_t)n * Kd + k];
    Y[tid] = relu ? fmaxf(acc, 0.f) : acc;
}

// =====================================================================
// GAGA: att = softmax_n(G[b,n,:].h[b,:]); out[b,:] = sum_n att_n G[b,n,:]
// =====================================================================
__global__ void gaga_kernel(const float* __restrict__ G,
                            const float* __restrict__ hvec,
                            float* __restrict__ out) {
    __shared__ float t[NN];
    __shared__ float pr[128];
    const int b = blockIdx.x;
    const int tid = threadIdx.x;                 // 128 threads
    const float* Gb = G + (size_t)b * NN * FOUT;
    const float* hb = hvec + b * FOUT;

    for (int n = tid; n < NN; n += 128) {
        const float4* gr = (const float4*)(Gb + (size_t)n * FOUT);
        const float4* hr = (const float4*)hb;
        float d = 0.f;
        for (int k = 0; k < FOUT / 4; ++k) {
            const float4 g = gr[k], hv = hr[k];
            d += g.x * hv.x + g.y * hv.y + g.z * hv.z + g.w * hv.w;
        }
        t[n] = d;
    }
    __syncthreads();
    float lm = -3.4e38f;
    for (int n = tid; n < NN; n += 128) lm = fmaxf(lm, t[n]);
    pr[tid] = lm;
    __syncthreads();
    if (tid == 0) { float m2 = pr[0]; for (int k = 1; k < 128; ++k) m2 = fmaxf(m2, pr[k]); pr[0] = m2; }
    __syncthreads();
    const float mxv = pr[0];
    __syncthreads();
    float ls = 0.f;
    for (int n = tid; n < NN; n += 128) ls += __expf(t[n] - mxv);
    pr[tid] = ls;
    __syncthreads();
    if (tid == 0) { float s = 0.f; for (int k = 0; k < 128; ++k) s += pr[k]; pr[0] = s; }
    __syncthreads();
    const float inv = 1.f / pr[0];
    for (int n = tid; n < NN; n += 128) t[n] = __expf(t[n] - mxv) * inv;
    __syncthreads();
    float acc = 0.f;
    for (int n = 0; n < NN; ++n) acc += t[n] * Gb[(size_t)n * FOUT + tid];
    out[b * FOUT + tid] = acc;
}

// =====================================================================
// LED + log_softmax: one block (256 threads) per batch.
// =====================================================================
__global__ void led_kernel(const float* __restrict__ ha, const float* __restrict__ gga,
                           const float* __restrict__ hb, const float* __restrict__ ggb,
                           const float* __restrict__ convW, const float* __restrict__ convb,
                           const float* __restrict__ W1, const float* __restrict__ b1,
                           const float* __restrict__ W2, const float* __restrict__ b2,
                           float* __restrict__ out) {
    __shared__ float ea[256], eb[256], xin[512], xv[128], lg[2];
    const int b = blockIdx.x;
    const int tid = threadIdx.x;                 // 256 threads
    if (tid < 128) {
        ea[tid] = ha[b * 128 + tid];
        eb[tid] = hb[b * 128 + tid];
    } else {
        ea[tid] = gga[b * 128 + tid - 128];
        eb[tid] = ggb[b * 128 + tid - 128];
    }
    __syncthreads();
    {
        float ca = convb[tid], cb = convb[tid];
        for (int k = 0; k < 256; ++k) {
            const float w = convW[(size_t)tid * 256 + k];
            ca += w * ea[k];
            cb += w * eb[k];
        }
        xin[tid] = fmaxf(ca, cb);                // maxpool over length-2 stack
        xin[256 + tid] = ea[tid] - eb[tid];
    }
    __syncthreads();
    if (tid < 128) {
        float acc = b1[tid];
        for (int k = 0; k < 512; ++k) acc += xin[k] * W1[(size_t)tid * 512 + k];
        xv[tid] = fmaxf(acc, 0.f);
    }
    __syncthreads();
    if (tid < 2) {
        float acc = b2[tid];
        for (int k = 0; k < 128; ++k) acc += xv[k] * W2[tid * 128 + k];
        lg[tid] = acc;
    }
    __syncthreads();
    if (tid == 0) {
        const float mm = fmaxf(lg[0], lg[1]);
        const float lse = mm + __logf(__expf(lg[0] - mm) + __expf(lg[1] - mm));
        out[b * 2 + 0] = lg[0] - lse;
        out[b * 2 + 1] = lg[1] - lse;
    }
}

// =====================================================================
extern "C" void kernel_launch(void* const* d_in, const int* in_sizes, int n_in,
                              void* d_out, int out_size, void* d_ws, size_t ws_size,
                              hipStream_t stream) {
    const float* a     = (const float*)d_in[0];
    const float* bio_a = (const float*)d_in[1];
    const int*   Aadj  = (const int*)d_in[2];
    const float* bvec  = (const float*)d_in[3];
    const float* bio_b = (const float*)d_in[4];
    const int*   Badj  = (const int*)d_in[5];
    const float* initW = (const float*)d_in[6];
    const float* initb = (const float*)d_in[7];
    const float* projW = (const float*)d_in[8];
    const float* attw  = (const float*)d_in[9];
    const float* attb  = (const float*)d_in[10];
    const float* ggeW1 = (const float*)d_in[11];
    const float* ggeb1 = (const float*)d_in[12];
    const float* ggeW2 = (const float*)d_in[13];
    const float* ggeb2 = (const float*)d_in[14];
    const float* convW = (const float*)d_in[15];
    const float* convb = (const float*)d_in[16];
    const float* ledW1 = (const float*)d_in[17];
    const float* ledb1 = (const float*)d_in[18];
    const float* ledW2 = (const float*)d_in[19];
    const float* ledb2 = (const float*)d_in[20];
    (void)in_sizes; (void)n_in; (void)out_size; (void)ws_size;

    float* ws = (float*)d_ws;
    const size_t BIG = (size_t)BT * NN * POUT;      // 1,048,576 floats
    const size_t RS  = (size_t)BT * HEAD * NN;      // 32,768 floats
    float* xa  = ws;
    float* xb  = xa  + BIG;
    float* xha = xb  + BIG;
    float* xhb = xha + BIG;
    float* Ga  = xhb + BIG;
    float* Gb  = Ga  + BIG;
    float* s1a = Gb  + BIG;
    float* s2a = s1a + RS;
    float* mxa = s2a + RS;
    float* ila = mxa + RS;
    float* s1b = ila + RS;
    float* s2b = s1b + RS;
    float* mxb = s2b + RS;
    float* ilb = mxb + RS;
    float* h1a = ilb + RS;          // [4,128]
    float* hha = h1a + BT * 128;
    float* h1b = hha + BT * 128;
    float* hhb = h1b + BT * 128;
    float* gga = hhb + BT * 128;
    float* ggb = gga + BT * 128;

    const int Mrows = BT * NN;      // 8192

    // 1) x = bio @ initW^T + initb   (bf16 WMMA)
    gemm_bf16_wmma<<<dim3(Mrows / 16, POUT / 64), 128, 0, stream>>>(bio_a, initW, initb, xa, Mrows, PIN, POUT, 0);
    gemm_bf16_wmma<<<dim3(Mrows / 16, POUT / 64), 128, 0, stream>>>(bio_b, initW, initb, xb, Mrows, PIN, POUT, 0);
    // 2) xh = x @ projW^T            (bf16 WMMA)
    gemm_bf16_wmma<<<dim3(Mrows / 16, POUT / 64), 128, 0, stream>>>(xa, projW, nullptr, xha, Mrows, POUT, POUT, 0);
    gemm_bf16_wmma<<<dim3(Mrows / 16, POUT / 64), 128, 0, stream>>>(xb, projW, nullptr, xhb, Mrows, POUT, POUT, 0);
    // 3) attention logits s1/s2
    s12_kernel<<<(BT * HEAD * NN) / 256, 256, 0, stream>>>(xha, attw, s1a, s2a);
    s12_kernel<<<(BT * HEAD * NN) / 256, 256, 0, stream>>>(xhb, attw, s1b, s2b);
    // 4) softmax row stats (flash pass 1)
    rowstats_kernel<<<(BT * HEAD * NN) / 8, 256, 0, stream>>>(Aadj, s1a, s2a, attb, mxa, ila);
    rowstats_kernel<<<(BT * HEAD * NN) / 8, 256, 0, stream>>>(Badj, s1b, s2b, attb, mxb, ilb);
    // 5) attention apply + relu + residual (flash pass 2, WMMA + TDM)
    attn_apply_kernel<<<BT * HEAD * (NN / 16), 128, 0, stream>>>(Aadj, s1a, s2a, attb, mxa, ila, xha, xa, Ga);
    attn_apply_kernel<<<BT * HEAD * (NN / 16), 128, 0, stream>>>(Badj, s1b, s2b, attb, mxb, ilb, xhb, xb, Gb);
    // 6) GGE (tiny)
    gemm_small<<<(BT * 128 + 255) / 256, 256, 0, stream>>>(a,    ggeW1, ggeb1, h1a, BT, FIN, 128, 1);
    gemm_small<<<(BT * FOUT + 255) / 256, 256, 0, stream>>>(h1a, ggeW2, ggeb2, hha, BT, 128, FOUT, 1);
    gemm_small<<<(BT * 128 + 255) / 256, 256, 0, stream>>>(bvec, ggeW1, ggeb1, h1b, BT, FIN, 128, 1);
    gemm_small<<<(BT * FOUT + 255) / 256, 256, 0, stream>>>(h1b, ggeW2, ggeb2, hhb, BT, 128, FOUT, 1);
    // 7) GAGA
    gaga_kernel<<<BT, 128, 0, stream>>>(Ga, hha, gga);
    gaga_kernel<<<BT, 128, 0, stream>>>(Gb, hhb, ggb);
    // 8) LED + log_softmax
    led_kernel<<<BT, 256, 0, stream>>>(hha, gga, hhb, ggb, convW, convb,
                                       ledW1, ledb1, ledW2, ledb2, (float*)d_out);
}
